// LocalAttention_51917564674243
// MI455X (gfx1250) — matrix operand order repaired and checked by reading, not compiled, not run on hardware
//
#include <hip/hip_runtime.h>
#include <hip/hip_bf16.h>
#include <math.h>
#include <stdint.h>

// ---------------------------------------------------------------------------
// T5 local attention for MI455X (gfx1250), wave32 + WMMA bf16.
// T=4096, B=4, C=1024, H=16, D=64, L=128, nb=32.  ~163 GFLOP, L2-resident
// -> compute bound -> everything through v_wmma_f32_16x16x32_bf16.
// All LDS fragment loads are 2x ds_load_b128; staging is b128 copies; the
// attention K/V staging uses CDNA5 async global->LDS (ASYNCcnt) copies.
// ---------------------------------------------------------------------------

typedef __bf16 v16bf __attribute__((ext_vector_type(16)));
typedef __bf16 v8bf  __attribute__((ext_vector_type(8)));
typedef __bf16 v4bf  __attribute__((ext_vector_type(4)));
typedef float  v8f   __attribute__((ext_vector_type(8)));
typedef float  v4f   __attribute__((ext_vector_type(4)));

#define EMBED  1024
#define HEADS  16
#define HDIM   64
#define BLKLEN 128
#define TLEN   4096
#define BATCH  4
#define NBLK   32
#define MROWS  (BATCH * TLEN)        // 16384
#define WT_STRIDE 72                 // transposed weight tile k-stride (padded)
#define JG 24                        // 384/16 key tiles
#define QT 8                         // 128/16 query tiles

__device__ __forceinline__ v8f zero8() {
  v8f z;
#pragma unroll
  for (int i = 0; i < 8; ++i) z[i] = 0.0f;
  return z;
}
__device__ __forceinline__ v8bf zero8bf() {
  v8bf z;
#pragma unroll
  for (int i = 0; i < 8; ++i) z[i] = (__bf16)0.0f;
  return z;
}

__device__ __forceinline__ v8f wmma_bf16(v16bf a, v16bf b, v8f c) {
  return __builtin_amdgcn_wmma_f32_16x16x32_bf16(
      false, a, false, b, (short)0, c, false, false);
}

__device__ __forceinline__ v16bf cat8(v8bf r0, v8bf r1) {
  return __builtin_shufflevector(r0, r1, 0, 1, 2, 3, 4, 5, 6, 7,
                                 8, 9, 10, 11, 12, 13, 14, 15);
}

// A-fragment (16x32 bf16) from row-major storage (stride mult of 8, 16B base).
__device__ __forceinline__ v16bf frag_a(const __bf16* p, int stride) {
  const int lane = threadIdx.x & 31;
  const __bf16* q = p + (lane & 15) * stride + 8 * (lane >> 4);
  return cat8(*(const v8bf*)(q), *(const v8bf*)(q + 16));
}

// B-fragment (32x16 bf16) from N-major storage: element(k,n) at p[n*stride+k].
__device__ __forceinline__ v16bf frag_b_t(const __bf16* p, int stride) {
  const int lane = threadIdx.x & 31;
  const __bf16* q = p + (lane & 15) * stride + 8 * (lane >> 4);
  return cat8(*(const v8bf*)(q), *(const v8bf*)(q + 16));
}

__device__ __forceinline__ float redmax16(float v) {
#pragma unroll
  for (int m = 1; m < 16; m <<= 1) v = fmaxf(v, __shfl_xor(v, m, 32));
  return v;
}
__device__ __forceinline__ float redsum16(float v) {
#pragma unroll
  for (int m = 1; m < 16; m <<= 1) v += __shfl_xor(v, m, 32);
  return v;
}

// CDNA5 async global->LDS copy (16B per lane), tracked by ASYNCcnt.
// vdst operand = LDS byte address (low 32 bits of the flat shared address).
__device__ __forceinline__ void async_copy_b128(void* lds, const void* gptr) {
  const unsigned l = (unsigned)(uintptr_t)lds;
  const unsigned long long g = (unsigned long long)(uintptr_t)gptr;
  asm volatile("global_load_async_to_lds_b128 %0, %1, off"
               :: "v"(l), "v"(g) : "memory");
}
__device__ __forceinline__ void wait_asynccnt0() {
  asm volatile("s_wait_asynccnt 0x0" ::: "memory");
}

// ---------------------------------------------------------------------------
// Kernel 0: weight transpose+convert: W (fp32, [k][n]) -> Wt (bf16, [n][k]).
// ---------------------------------------------------------------------------
__global__ __launch_bounds__(256) void la_wt_kernel(
    const float* __restrict__ W0, const float* __restrict__ W1,
    const float* __restrict__ W2, const float* __restrict__ W3,
    __bf16* __restrict__ T0, __bf16* __restrict__ T1,
    __bf16* __restrict__ T2, __bf16* __restrict__ T3) {
  __shared__ __align__(16) __bf16 sT[64 * WT_STRIDE];
  const float* W = (blockIdx.z == 0) ? W0 : (blockIdx.z == 1) ? W1
                   : (blockIdx.z == 2) ? W2 : W3;
  __bf16* T = (blockIdx.z == 0) ? T0 : (blockIdx.z == 1) ? T1
              : (blockIdx.z == 2) ? T2 : T3;
  const int k0 = blockIdx.x * 64, n0 = blockIdx.y * 64;
  const int tid = threadIdx.x;
#pragma unroll
  for (int c4 = 0; c4 < 4; ++c4) {
    const int lin = c4 * 1024 + tid * 4;
    const int kk = lin >> 6, nn = lin & 63;
    const v4f w = *(const v4f*)(&W[(size_t)(k0 + kk) * EMBED + n0 + nn]);
#pragma unroll
    for (int e = 0; e < 4; ++e) sT[(nn + e) * WT_STRIDE + kk] = (__bf16)w[e];
  }
  __syncthreads();
#pragma unroll
  for (int c2 = 0; c2 < 2; ++c2) {
    const int lin = c2 * 2048 + tid * 8;
    const int nn = lin >> 6, kk = lin & 63;
    *(v8bf*)(&T[(size_t)(n0 + nn) * EMBED + k0 + kk]) =
        *(const v8bf*)(&sT[nn * WT_STRIDE + kk]);
  }
}

// ---------------------------------------------------------------------------
// Kernel 1: relative position bias in C-fragment layout:
// biasF[(((h*8 + qtile)*24 + jtile)*32 + lane)*8 + r]
// element = bias(qrow = qtile*16 + r + 8*(lane>>4), key = jtile*16 + (lane&15))
// ---------------------------------------------------------------------------
__global__ void la_bias_kernel(const float* __restrict__ table,
                               float* __restrict__ biasF) {
  const int g = blockIdx.x * blockDim.x + threadIdx.x;
  if (g >= HEADS * QT * JG * 32) return;
  const int lane = g & 31;
  const int jg = (g >> 5) % JG;
  const int w = ((g >> 5) / JG) % QT;
  const int h = (g >> 5) / (JG * QT);
  const int hi = lane >> 4, n = lane & 15;
  const int key = jg * 16 + n;
  float* outp = biasF + (size_t)g * 8;
#pragma unroll
  for (int r = 0; r < 8; ++r) {
    const int qrow = w * 16 + r + 8 * hi;
    const int rp = key - (BLKLEN + qrow);
    const int base = (rp > 0) ? 16 : 0;
    const int arp = rp < 0 ? -rp : rp;
    int bucket;
    if (arp < 8) {
      bucket = base + arp;
    } else {
      const float f = __logf((float)arp * 0.125f) * (8.0f / 2.7725887222397811f);
      int big = 8 + (int)f;
      if (big > 15) big = 15;
      bucket = base + big;
    }
    outp[r] = table[bucket * HEADS + h];
  }
}

// ---------------------------------------------------------------------------
// Kernel 2: fused QKV projection.  Block tile 128(M) x 64(N), K chunks of 64.
// N-tile == one head (64 cols), so V is stored transposed per head:
// Vt[((b*16+h)*64 + d)*4096 + t].  Q is pre-scaled by 0.125.
// ---------------------------------------------------------------------------
__global__ __launch_bounds__(256) void la_qkv_kernel(
    const float* __restrict__ hs,
    const __bf16* __restrict__ Wqt, const __bf16* __restrict__ Wkt,
    const __bf16* __restrict__ Wvt,
    __bf16* __restrict__ Qb, __bf16* __restrict__ Kb, __bf16* __restrict__ Vt) {
  __shared__ __align__(16) __bf16 sA[128 * 64];            // 16 KB
  __shared__ __align__(16) __bf16 sWq[64 * WT_STRIDE];     // 9 KB each
  __shared__ __align__(16) __bf16 sWk[64 * WT_STRIDE];
  __shared__ __align__(16) __bf16 sWv[64 * WT_STRIDE];

  const int m0 = blockIdx.x * 128;
  const int n0 = blockIdx.y * 64;
  const int tid = threadIdx.x;
  const int wave = tid >> 5;
  const int lane = tid & 31;

  v8f accq[4], acck[4], accv[4];
#pragma unroll
  for (int j = 0; j < 4; ++j) { accq[j] = zero8(); acck[j] = zero8(); accv[j] = zero8(); }

  for (int kc = 0; kc < EMBED; kc += 64) {
    // stage A (fp32 -> bf16), row-major, float4 loads + b64 stores
#pragma unroll
    for (int c4 = 0; c4 < 8; ++c4) {
      const int lin = c4 * 1024 + tid * 4;
      const int r = lin >> 6, cc = lin & 63;
      const int m = m0 + r;
      const int t = m & (TLEN - 1), b = m >> 12;
      const v4f w = *(const v4f*)(&hs[((size_t)t * BATCH + b) * EMBED + kc + cc]);
      v4bf o;
#pragma unroll
      for (int e = 0; e < 4; ++e) o[e] = (__bf16)w[e];
      *(v4bf*)(&sA[r * 64 + cc]) = o;
    }
    // stage pre-transposed weight tiles: pure b128 copies
#pragma unroll
    for (int c2 = 0; c2 < 2; ++c2) {
      const int lin = c2 * 2048 + tid * 8;
      const int nn = lin >> 6, kk = lin & 63;
      const size_t off = (size_t)(n0 + nn) * EMBED + kc + kk;
      const int li = nn * WT_STRIDE + kk;
      *(v8bf*)(&sWq[li]) = *(const v8bf*)(&Wqt[off]);
      *(v8bf*)(&sWk[li]) = *(const v8bf*)(&Wkt[off]);
      *(v8bf*)(&sWv[li]) = *(const v8bf*)(&Wvt[off]);
    }
    __syncthreads();

    const v16bf a0 = frag_a(&sA[(wave * 16) * 64], 64);
    const v16bf a1 = frag_a(&sA[(wave * 16) * 64 + 32], 64);
#pragma unroll
    for (int j = 0; j < 4; ++j) {
      const __bf16* pq = &sWq[(16 * j) * WT_STRIDE];
      const __bf16* pk = &sWk[(16 * j) * WT_STRIDE];
      const __bf16* pv = &sWv[(16 * j) * WT_STRIDE];
      const v16bf bq0 = frag_b_t(pq, WT_STRIDE);
      const v16bf bq1 = frag_b_t(pq + 32, WT_STRIDE);
      const v16bf bk0 = frag_b_t(pk, WT_STRIDE);
      const v16bf bk1 = frag_b_t(pk + 32, WT_STRIDE);
      const v16bf bv0 = frag_b_t(pv, WT_STRIDE);
      const v16bf bv1 = frag_b_t(pv + 32, WT_STRIDE);
      accq[j] = wmma_bf16(a0, bq0, accq[j]);
      accq[j] = wmma_bf16(a1, bq1, accq[j]);
      acck[j] = wmma_bf16(a0, bk0, acck[j]);
      acck[j] = wmma_bf16(a1, bk1, acck[j]);
      accv[j] = wmma_bf16(a0, bv0, accv[j]);
      accv[j] = wmma_bf16(a1, bv1, accv[j]);
    }
    __syncthreads();
  }

  const int hi = lane >> 4, n = lane & 15;
  const int h = n0 >> 6;               // N-tile == head
  const int b = m0 >> 12, t0 = m0 & (TLEN - 1);

  // ---- Q: stage row-major in sA, cooperative b128 store (scaled) ----
#pragma unroll
  for (int j = 0; j < 4; ++j)
#pragma unroll
    for (int r = 0; r < 8; ++r)
      sA[(wave * 16 + r + 8 * hi) * 64 + 16 * j + n] = (__bf16)(accq[j][r] * 0.125f);
  __syncthreads();
#pragma unroll
  for (int c4 = 0; c4 < 4; ++c4) {
    const int lin = c4 * 2048 + tid * 8;
    const int r = lin >> 6, cc = lin & 63;
    *(v8bf*)(&Qb[(size_t)(m0 + r) * EMBED + n0 + cc]) = *(const v8bf*)(&sA[r * 64 + cc]);
  }
  __syncthreads();

  // ---- K: same, unscaled ----
#pragma unroll
  for (int j = 0; j < 4; ++j)
#pragma unroll
    for (int r = 0; r < 8; ++r)
      sA[(wave * 16 + r + 8 * hi) * 64 + 16 * j + n] = (__bf16)acck[j][r];
  __syncthreads();
#pragma unroll
  for (int c4 = 0; c4 < 4; ++c4) {
    const int lin = c4 * 2048 + tid * 8;
    const int r = lin >> 6, cc = lin & 63;
    *(v8bf*)(&Kb[(size_t)(m0 + r) * EMBED + n0 + cc]) = *(const v8bf*)(&sA[r * 64 + cc]);
  }
  __syncthreads();

  // ---- V: stage transposed [d][t] in sA, cooperative b128 store to Vt ----
#pragma unroll
  for (int j = 0; j < 4; ++j)
#pragma unroll
    for (int r = 0; r < 8; ++r)
      sA[(16 * j + n) * 128 + (wave * 16 + r + 8 * hi)] = (__bf16)accv[j][r];
  __syncthreads();
#pragma unroll
  for (int c4 = 0; c4 < 4; ++c4) {
    const int lin = c4 * 2048 + tid * 8;
    const int d = lin >> 7, tt = lin & 127;
    *(v8bf*)(&Vt[((size_t)(b * HEADS + h) * HDIM + d) * TLEN + t0 + tt]) =
        *(const v8bf*)(&sA[d * 128 + tt]);
  }
}

// ---------------------------------------------------------------------------
// Kernel 3: blocked attention, flash-style over the 3 key blocks.
// One workgroup per (b, blk, h); 8 waves x 16 query rows.
// K/V chunks staged with async global->LDS (ASYNCcnt); sK row-major
// (N-major for Q·K^T), sVt from pre-transposed Vt, sP row-major (A for P·V).
// LDS = 16 + 16 + 32 = 64 KB.
// ---------------------------------------------------------------------------
__global__ __launch_bounds__(256) void la_attn_kernel(
    const __bf16* __restrict__ Qb, const __bf16* __restrict__ Kb,
    const __bf16* __restrict__ Vt, const float* __restrict__ biasF,
    __bf16* __restrict__ Ctx) {
  __shared__ __align__(16) __bf16 sK[128 * 64];    // [key][d]   16 KB
  __shared__ __align__(16) __bf16 sVt[64 * 128];   // [d][key]   16 KB
  __shared__ __align__(16) __bf16 sP[128 * 128];   // [q][key]   32 KB

  const int bid = blockIdx.x;           // b*512 + blk*16 + h
  const int h = bid & 15;
  const int blk = (bid >> 4) & 31;
  const int b = bid >> 9;
  const int tid = threadIdx.x;
  const int wave = tid >> 5;
  const int lane = tid & 31;
  const int hi = lane >> 4, n = lane & 15;

  const __bf16* qbase =
      Qb + ((size_t)(b * TLEN + blk * BLKLEN + wave * 16)) * EMBED + h * HDIM;
  const v16bf qa0 = frag_a(qbase, EMBED);
  const v16bf qa1 = frag_a(qbase + 32, EMBED);

  float mrow[8], lrow[8];
  v8f o[4];
#pragma unroll
  for (int r = 0; r < 8; ++r) { mrow[r] = -1e30f; lrow[r] = 0.0f; }
#pragma unroll
  for (int nt = 0; nt < 4; ++nt) o[nt] = zero8();

  const float* bF = biasF + ((size_t)((h * QT + wave) * JG) * 32 + lane) * 8;

  for (int c = 0; c < 3; ++c) {
    const int gb = blk - 1 + c;
    const bool valid = (gb >= 0) && (gb < NBLK);
    if (valid) {
      // async global->LDS copies, 16B per lane per instruction
#pragma unroll
      for (int c4 = 0; c4 < 4; ++c4) {
        const int lin = c4 * 2048 + tid * 8;
        const int r = lin >> 6, dd = lin & 63;
        async_copy_b128(&sK[r * 64 + dd],
                        &Kb[(size_t)(b * TLEN + gb * BLKLEN + r) * EMBED + h * HDIM + dd]);
        const int d = lin >> 7, rr = lin & 127;
        async_copy_b128(&sVt[d * 128 + rr],
                        &Vt[((size_t)(b * HEADS + h) * HDIM + d) * TLEN + gb * BLKLEN + rr]);
      }
      wait_asynccnt0();
    } else {
      const v8bf z = zero8bf();
#pragma unroll
      for (int c4 = 0; c4 < 4; ++c4) {
        const int lin = c4 * 2048 + tid * 8;
        *(v8bf*)(&sK[lin]) = z;
        *(v8bf*)(&sVt[lin]) = z;
      }
    }
    __syncthreads();

    // scores for 8 column tiles of this 128-key chunk (+ fragment-layout bias)
    v8f s[8];
#pragma unroll
    for (int j = 0; j < 8; ++j) {
      const v16bf kb0 = frag_b_t(&sK[(16 * j) * 64], 64);
      const v16bf kb1 = frag_b_t(&sK[(16 * j) * 64 + 32], 64);
      v8f acc = zero8();
      acc = wmma_bf16(qa0, kb0, acc);
      acc = wmma_bf16(qa1, kb1, acc);
      const v8f bb = *(const v8f*)(bF + (size_t)(c * 8 + j) * (32 * 8));
      s[j] = acc + bb;
    }

    // online softmax update (row = r + 8*hi, shared by 16 lanes)
#pragma unroll
    for (int r = 0; r < 8; ++r) {
      float cm = -1e30f;
#pragma unroll
      for (int j = 0; j < 8; ++j) cm = fmaxf(cm, s[j][r]);
      cm = redmax16(cm);
      const float mn = fmaxf(mrow[r], cm);
      const float corr = __expf(mrow[r] - mn);
      mrow[r] = mn;
      float ps = 0.0f;
#pragma unroll
      for (int j = 0; j < 8; ++j) {
        const float p = __expf(s[j][r] - mn);
        ps += p;
        s[j][r] = p;
      }
      ps = redsum16(ps);
      lrow[r] = lrow[r] * corr + ps;
#pragma unroll
      for (int nt = 0; nt < 4; ++nt) o[nt][r] *= corr;
    }

    // stage probabilities (bf16) -- each wave owns its 16 rows of sP
#pragma unroll
    for (int j = 0; j < 8; ++j)
#pragma unroll
      for (int r = 0; r < 8; ++r)
        sP[(wave * 16 + r + 8 * hi) * 128 + (16 * j + n)] = (__bf16)s[j][r];

    // O += P(16x128) * V(128x64)
#pragma unroll
    for (int kt = 0; kt < 4; ++kt) {
      const v16bf pa = frag_a(&sP[(wave * 16) * 128 + kt * 32], 128);
      const v16bf vb0 = frag_b_t(&sVt[0 * 128 + kt * 32], 128);
      const v16bf vb1 = frag_b_t(&sVt[16 * 128 + kt * 32], 128);
      const v16bf vb2 = frag_b_t(&sVt[32 * 128 + kt * 32], 128);
      const v16bf vb3 = frag_b_t(&sVt[48 * 128 + kt * 32], 128);
      o[0] = wmma_bf16(pa, vb0, o[0]);
      o[1] = wmma_bf16(pa, vb1, o[1]);
      o[2] = wmma_bf16(pa, vb2, o[2]);
      o[3] = wmma_bf16(pa, vb3, o[3]);
    }
    __syncthreads();
  }

  // normalize, stage in sP ([q][64]), cooperative b128 store to Ctx
#pragma unroll
  for (int nt = 0; nt < 4; ++nt)
#pragma unroll
    for (int r = 0; r < 8; ++r)
      sP[(wave * 16 + r + 8 * hi) * 64 + nt * 16 + n] = (__bf16)(o[nt][r] / lrow[r]);
  __syncthreads();
#pragma unroll
  for (int c4 = 0; c4 < 4; ++c4) {
    const int lin = c4 * 2048 + tid * 8;
    const int r = lin >> 6, cc = lin & 63;
    *(v8bf*)(&Ctx[(size_t)(b * TLEN + blk * BLKLEN + r) * EMBED + h * HDIM + cc]) =
        *(const v8bf*)(&sP[r * 64 + cc]);
  }
}

// ---------------------------------------------------------------------------
// Kernel 4: output projection.  out[t,b,n] = sum_k Ctx[m,k]*Wo[k,n], m=b*T+t.
// ---------------------------------------------------------------------------
__global__ __launch_bounds__(256) void la_out_kernel(
    const __bf16* __restrict__ Ctx, const __bf16* __restrict__ Wot,
    float* __restrict__ out) {
  __shared__ __align__(16) __bf16 sA[128 * 64];          // 16 KB
  __shared__ __align__(16) __bf16 sW[64 * WT_STRIDE];    // 9 KB
  __shared__ __align__(16) float  sOut[128 * 64];        // 32 KB

  const int m0 = blockIdx.x * 128;
  const int n0 = blockIdx.y * 64;
  const int tid = threadIdx.x;
  const int wave = tid >> 5;
  const int lane = tid & 31;

  v8f acc[4];
#pragma unroll
  for (int j = 0; j < 4; ++j) acc[j] = zero8();

  for (int kc = 0; kc < EMBED; kc += 64) {
#pragma unroll
    for (int c4 = 0; c4 < 4; ++c4) {
      const int lin = c4 * 2048 + tid * 8;
      const int r = lin >> 6, cc = lin & 63;
      *(v8bf*)(&sA[r * 64 + cc]) =
          *(const v8bf*)(&Ctx[(size_t)(m0 + r) * EMBED + kc + cc]);
    }
#pragma unroll
    for (int c2 = 0; c2 < 2; ++c2) {
      const int lin = c2 * 2048 + tid * 8;
      const int nn = lin >> 6, kk = lin & 63;
      *(v8bf*)(&sW[nn * WT_STRIDE + kk]) =
          *(const v8bf*)(&Wot[(size_t)(n0 + nn) * EMBED + kc + kk]);
    }
    __syncthreads();

    const v16bf a0 = frag_a(&sA[(wave * 16) * 64], 64);
    const v16bf a1 = frag_a(&sA[(wave * 16) * 64 + 32], 64);
#pragma unroll
    for (int j = 0; j < 4; ++j) {
      const v16bf b0 = frag_b_t(&sW[(16 * j) * WT_STRIDE], WT_STRIDE);
      const v16bf b1 = frag_b_t(&sW[(16 * j) * WT_STRIDE + 32], WT_STRIDE);
      acc[j] = wmma_bf16(a0, b0, acc[j]);
      acc[j] = wmma_bf16(a1, b1, acc[j]);
    }
    __syncthreads();
  }

  // stage fp32 tile in LDS, cooperative float4 store (out layout (T,B,C))
  const int hi = lane >> 4, n = lane & 15;
#pragma unroll
  for (int j = 0; j < 4; ++j)
#pragma unroll
    for (int r = 0; r < 8; ++r)
      sOut[(wave * 16 + r + 8 * hi) * 64 + 16 * j + n] = acc[j][r];
  __syncthreads();
#pragma unroll
  for (int c4 = 0; c4 < 8; ++c4) {
    const int lin = c4 * 1024 + tid * 4;
    const int r = lin >> 6, cc = lin & 63;
    const int m = m0 + r;
    const int t = m & (TLEN - 1), b = m >> 12;
    *(v4f*)(&out[((size_t)t * BATCH + b) * EMBED + n0 + cc]) =
        *(const v4f*)(&sOut[r * 64 + cc]);
  }
}

// ---------------------------------------------------------------------------
extern "C" void kernel_launch(void* const* d_in, const int* in_sizes, int n_in,
                              void* d_out, int out_size, void* d_ws,
                              size_t ws_size, hipStream_t stream) {
  (void)in_sizes; (void)n_in; (void)out_size; (void)ws_size;

  const float* hs    = (const float*)d_in[0];
  const float* Wq    = (const float*)d_in[1];
  const float* Wk    = (const float*)d_in[2];
  const float* Wv    = (const float*)d_in[3];
  const float* Wo    = (const float*)d_in[4];
  const float* table = (const float*)d_in[5];
  float* out = (float*)d_out;

  const size_t SZ = (size_t)MROWS * EMBED * sizeof(__bf16);  // 32 MB each
  const size_t WSZ = (size_t)EMBED * EMBED * sizeof(__bf16); // 2 MB each
  char* ws = (char*)d_ws;
  __bf16* Qb   = (__bf16*)(ws + 0 * SZ);
  __bf16* Kb   = (__bf16*)(ws + 1 * SZ);
  __bf16* Vt   = (__bf16*)(ws + 2 * SZ);   // [(b*16+h)*64+d][t]
  __bf16* Ctx  = (__bf16*)(ws + 3 * SZ);
  float*  biasF = (float*)(ws + 4 * SZ);   // 3 MB, fragment layout
  __bf16* Wqt  = (__bf16*)(ws + 4 * SZ + 4 * 1024 * 1024);
  __bf16* Wkt  = (__bf16*)((char*)Wqt + WSZ);
  __bf16* Wvt  = (__bf16*)((char*)Wkt + WSZ);
  __bf16* Wot  = (__bf16*)((char*)Wvt + WSZ);

  dim3 gWt(EMBED / 64, EMBED / 64, 4);
  la_wt_kernel<<<gWt, 256, 0, stream>>>(Wq, Wk, Wv, Wo, Wqt, Wkt, Wvt, Wot);

  la_bias_kernel<<<(HEADS * QT * JG * 32 + 255) / 256, 256, 0, stream>>>(table, biasF);

  dim3 gGemm(MROWS / 128, EMBED / 64);
  la_qkv_kernel<<<gGemm, 256, 0, stream>>>(hs, Wqt, Wkt, Wvt, Qb, Kb, Vt);

  la_attn_kernel<<<BATCH * NBLK * HEADS, 256, 0, stream>>>(Qb, Kb, Vt, biasF, Ctx);

  la_out_kernel<<<gGemm, 256, 0, stream>>>(Ctx, Wot, out);
}
